// LabelTempBiasSelfAttn_42425686950378
// MI455X (gfx1250) — matrix operand order repaired and checked by reading, not compiled, not run on hardware
//
#include <hip/hip_runtime.h>
#include <hip/hip_bf16.h>
#include <math.h>

// ---------------------------------------------------------------------------
// LabelTempBiasSelfAttn for MI455X (gfx1250), wave32 + WMMA f16 path.
// V is stored transposed (channel-major) so every WMMA fragment load in the
// attention kernel is two 16B runs per lane (2x b128).
// ---------------------------------------------------------------------------

typedef __attribute__((ext_vector_type(16))) _Float16 v16h;
typedef __attribute__((ext_vector_type(8)))  _Float16 v8h;
typedef __attribute__((ext_vector_type(8)))  float    v8f;

#define D_MODEL   1024
#define NHEAD     16
#define HEAD_DIM  64
#define SEQ_T     2048
#define NBATCH    4
#define MTOT      (NBATCH * SEQ_T)   // 8192
#define TEMP_HIDDEN 32
#define BIAS_HIDDEN 64
#define TILE_M 128
#define TILE_N 64

__device__ __forceinline__ v8f wmma_f16(v16h a, v16h b, v8f c) {
  // D = A(16x32 f16) * B(32x16 f16) + C(16x16 f32)
  return __builtin_amdgcn_wmma_f32_16x16x32_f16(false, a, false, b, (short)0, c,
                                                false, false);
}

// A-fragment (16x32, f16), M rows striped over lanes, K over VGPRs.
// lane L: row = m0 + (L&15); grp = L>>4; halves: v0..3 -> K=grp*8+{0..7},
// v4..7 -> K=16+grp*8+{0..7}. Two contiguous 16B runs per lane.
__device__ __forceinline__ v16h load_A_frag(const _Float16* __restrict__ A,
                                            int lda, int m0, int k0, int lane) {
  int row = m0 + (lane & 15);
  int grp = lane >> 4;
  const _Float16* base = A + (size_t)row * lda + k0 + grp * 8;
  v8h lo = *reinterpret_cast<const v8h*>(base);
  v8h hi = *reinterpret_cast<const v8h*>(base + 16);
  v16h r;
#pragma unroll
  for (int i = 0; i < 8; ++i) { r[i] = lo[i]; r[8 + i] = hi[i]; }
  return r;
}

// B-fragment (32x16, f16) of W^T where W is row-major [N,K]:
// B[k][n] = W[n][k]. lane L: n = n0 + (L&15); K = (L>>4)*8 + {0..7}, 16 + (L>>4)*8 + {0..7}.
// => two 16B runs of W row n per lane (2x b128).
__device__ __forceinline__ v16h load_B_fragT(const _Float16* __restrict__ W,
                                             size_t ldw, int n0, int k0, int lane) {
  int n  = n0 + (lane & 15);
  int kb = k0 + (lane >> 4) * 8;
  const _Float16* base = W + (size_t)n * ldw + kb;
  v8h lo = *reinterpret_cast<const v8h*>(base);
  v8h hi = *reinterpret_cast<const v8h*>(base + 16);
  v16h r;
#pragma unroll
  for (int i = 0; i < 8; ++i) { r[i] = lo[i]; r[8 + i] = hi[i]; }
  return r;
}

// A-fragment from an LDS strip [16 rows x (64+pad) cols], stride in halves.
__device__ __forceinline__ v16h load_A_lds(const _Float16* __restrict__ L,
                                           int stride, int lane) {
  int row = lane & 15;
  int grp = lane >> 4;
  const _Float16* base = L + row * stride + grp * 8;
  v16h r;
#pragma unroll
  for (int i = 0; i < 8; ++i) { r[i] = base[i]; r[8 + i] = base[16 + i]; }
  return r;
}

// ---------------------------------------------------------------------------
__global__ void zero_f32(float* p, int n) {
  int i = blockIdx.x * blockDim.x + threadIdx.x;
  if (i < n) p[i] = 0.0f;
}

// ---------------------------------------------------------------------------
// Per-batch scalar MLPs. LayerNorm over a size-1 dim => normalized value is
// exactly 0 => hidden input == ln_bias (g multiplied by zero). Exact.
__global__ void prep_scalars(const float* __restrict__ labels,
                             const float* __restrict__ tau_ln_b,
                             const float* __restrict__ tau_w1,
                             const float* __restrict__ tau_b1,
                             const float* __restrict__ tau_w2,
                             const float* __restrict__ tau_b2,
                             const float* __restrict__ bias_ln_b,
                             const float* __restrict__ bias_w1,
                             const float* __restrict__ bias_b1,
                             const float* __restrict__ bias_w2,
                             const float* __restrict__ bias_b2,
                             float* __restrict__ qscale,
                             float* __restrict__ bscalar) {
  int b = threadIdx.x;
  if (b >= NBATCH) return;
  (void)labels;
  float h0 = tau_ln_b[0];
  float acc = tau_b2[0];
  for (int i = 0; i < TEMP_HIDDEN; ++i) {
    float h = fmaxf(h0 * tau_w1[i] + tau_b1[i], 0.0f);
    acc += h * tau_w2[i];
  }
  float sp  = (acc > 20.0f) ? acc : log1pf(__expf(acc));
  float tau = sp + 1e-3f;
  qscale[b] = rsqrtf(tau) * 0.125f;  // fold 1/sqrt(hd)=1/8 into Q
  float h0b = bias_ln_b[0];
  float accb = bias_b2[0];
  for (int i = 0; i < BIAS_HIDDEN; ++i) {
    float h = fmaxf(h0b * bias_w1[i] + bias_b1[i], 0.0f);
    accb += h * bias_w2[i];
  }
  bscalar[b] = accb;
}

// ---------------------------------------------------------------------------
// Per-row: x -> f16, key padding mask, per-key bias, den count.
__global__ void __launch_bounds__(256) prep_rows(const float* __restrict__ x,
                                                 _Float16* __restrict__ xh,
                                                 float* __restrict__ maskrow,
                                                 float* __restrict__ biaskey,
                                                 float* __restrict__ den,
                                                 const float* __restrict__ bscalar,
                                                 const float* __restrict__ pos_embed) {
  int row = blockIdx.x;              // 0 .. B*T-1
  int b = row / SEQ_T, t = row % SEQ_T;
  const float* xr = x + (size_t)row * D_MODEL;
  _Float16* xo = xh + (size_t)row * D_MODEL;
  float asum = 0.0f;
  for (int i = threadIdx.x; i < D_MODEL; i += 256) {
    float v = xr[i];
    asum += fabsf(v);
    xo[i] = (_Float16)v;
  }
  __shared__ float red[256];
  red[threadIdx.x] = asum;
  __syncthreads();
  for (int s = 128; s > 0; s >>= 1) {
    if (threadIdx.x < s) red[threadIdx.x] += red[threadIdx.x + s];
    __syncthreads();
  }
  if (threadIdx.x == 0) {
    bool kpm = red[0] <= 1e-6f;
    float m = kpm ? 0.0f : 1.0f;
    maskrow[row] = m;
    biaskey[row] = kpm ? -3.0e38f : bscalar[b] * pos_embed[t];
    atomicAdd(&den[b], m);
  }
}

// ---------------------------------------------------------------------------
__global__ void f2h(const float* __restrict__ in, _Float16* __restrict__ out, int n) {
  for (int i = blockIdx.x * blockDim.x + threadIdx.x; i < n;
       i += gridDim.x * blockDim.x)
    out[i] = (_Float16)in[i];
}

// ---------------------------------------------------------------------------
// C = A[M,K] @ W[N,K]^T via WMMA, f16 in / f32 acc / f16 out.
// block = 256 thr (8 waves), block tile 128x64, wave tile 32x32.
// transStore==0: C row-major [M,N].  transStore==1: C channel-major [N,MTOT]
// (each lane then stores 8 contiguous halves = one 16B v8h store).
__global__ void __launch_bounds__(256) gemm_proj(const _Float16* __restrict__ A,
                                                 const _Float16* __restrict__ W,
                                                 _Float16* __restrict__ C,
                                                 int N, int K,
                                                 const float* __restrict__ qscale,
                                                 int useScale, int transStore) {
  int wave = threadIdx.x >> 5;
  int lane = threadIdx.x & 31;
  int m0 = blockIdx.x * TILE_M + (wave >> 1) * 32;
  int n0 = blockIdx.y * TILE_N + (wave & 1) * 32;
  v8f acc[2][2] = {};
  for (int k0 = 0; k0 < K; k0 += 32) {
    if (k0 + 64 < K)
      __builtin_prefetch(A + (size_t)(m0 + (lane & 15)) * K + k0 + 64, 0, 1);
    v16h a0 = load_A_frag(A, K, m0, k0, lane);
    v16h a1 = load_A_frag(A, K, m0 + 16, k0, lane);
    v16h b0 = load_B_fragT(W, K, n0, k0, lane);
    v16h b1 = load_B_fragT(W, K, n0 + 16, k0, lane);
    acc[0][0] = wmma_f16(a0, b0, acc[0][0]);
    acc[0][1] = wmma_f16(a0, b1, acc[0][1]);
    acc[1][0] = wmma_f16(a1, b0, acc[1][0]);
    acc[1][1] = wmma_f16(a1, b1, acc[1][1]);
  }
  float sc = useScale ? qscale[(blockIdx.x * TILE_M) / SEQ_T] : 1.0f;
  if (transStore) {
#pragma unroll
    for (int i = 0; i < 2; ++i)
#pragma unroll
      for (int j = 0; j < 2; ++j) {
        int n  = n0 + j * 16 + (lane & 15);
        int mb = m0 + i * 16 + (lane >> 4) * 8;
        v8h hv;
#pragma unroll
        for (int v = 0; v < 8; ++v) hv[v] = (_Float16)(acc[i][j][v] * sc);
        *reinterpret_cast<v8h*>(C + (size_t)n * MTOT + mb) = hv;
      }
  } else {
#pragma unroll
    for (int i = 0; i < 2; ++i)
#pragma unroll
      for (int j = 0; j < 2; ++j) {
        int n  = n0 + j * 16 + (lane & 15);
        int mb = m0 + i * 16 + (lane >> 4) * 8;
#pragma unroll
        for (int v = 0; v < 8; ++v)
          C[(size_t)(mb + v) * N + n] = (_Float16)(acc[i][j][v] * sc);
      }
  }
}

// ---------------------------------------------------------------------------
// Flash attention + masked-mean reduction over queries.
// grid: (T/64, H, B); block: 128 thr = 4 waves; wave owns a 16-query strip.
// Key chunk = 64 (4 column tiles). V is channel-major Vt[ch][token].
__global__ void __launch_bounds__(128) attn_fused(const _Float16* __restrict__ Qh,
                                                  const _Float16* __restrict__ Kh,
                                                  const _Float16* __restrict__ Vt,
                                                  const float* __restrict__ biaskey,
                                                  const float* __restrict__ maskrow,
                                                  float* __restrict__ sAcc) {
  __shared__ __attribute__((aligned(16))) _Float16 pbuf[4][16 * 72];
  int b = blockIdx.z, h = blockIdx.y;
  int q0 = blockIdx.x * 64;
  int wave = threadIdx.x >> 5, lane = threadIdx.x & 31;
  int grp = lane >> 4;
  int qw = q0 + wave * 16;

  const _Float16* Qp = Qh + (size_t)b * SEQ_T * D_MODEL + h * HEAD_DIM;
  const _Float16* Kp = Kh + (size_t)b * SEQ_T * D_MODEL + h * HEAD_DIM;
  // Vt channel-major: Vt[ch][token], token = b*SEQ_T + t
  const _Float16* Vp = Vt + (size_t)(h * HEAD_DIM) * MTOT + (size_t)b * SEQ_T;
  const float* bk = biaskey + (size_t)b * SEQ_T;

  // Q strip fragments (hd=64 -> two 16x32 A fragments), kept in registers
  v16h qa0 = load_A_frag(Qp, D_MODEL, qw, 0, lane);
  v16h qa1 = load_A_frag(Qp, D_MODEL, qw, 32, lane);

  v8f Oacc[4] = {};
  float rowmax[8], rowsum[8];
#pragma unroll
  for (int v = 0; v < 8; ++v) { rowmax[v] = -1e30f; rowsum[v] = 0.0f; }

  for (int key0 = 0; key0 < SEQ_T; key0 += 64) {
    // S strip [16 x 64] = Q strip @ K^T  (4 column tiles, 2 K-steps each)
    v8f s[4];
#pragma unroll
    for (int j = 0; j < 4; ++j) {
      v8f z = {};
      v16h kb0 = load_B_fragT(Kp, D_MODEL, key0 + j * 16, 0, lane);
      v16h kb1 = load_B_fragT(Kp, D_MODEL, key0 + j * 16, 32, lane);
      s[j] = wmma_f16(qa0, kb0, z);
      s[j] = wmma_f16(qa1, kb1, s[j]);
    }

    // per-key bias (per column)
    float bj[4];
#pragma unroll
    for (int j = 0; j < 4; ++j) bj[j] = bk[key0 + j * 16 + (lane & 15)];
#pragma unroll
    for (int j = 0; j < 4; ++j)
#pragma unroll
      for (int v = 0; v < 8; ++v) s[j][v] += bj[j];

    // online softmax: row stats via 16-lane-group shuffle reductions
#pragma unroll
    for (int v = 0; v < 8; ++v) {
      float mv = fmaxf(fmaxf(s[0][v], s[1][v]), fmaxf(s[2][v], s[3][v]));
#pragma unroll
      for (int off = 1; off < 16; off <<= 1)
        mv = fmaxf(mv, __shfl_xor(mv, off));
      float mold = rowmax[v];
      float mnew = fmaxf(fmaxf(mold, mv), -1e30f);
      rowmax[v] = mnew;
      float scl = __expf(mold - mnew);
      float ps = 0.0f;
#pragma unroll
      for (int j = 0; j < 4; ++j) {
        s[j][v] = __expf(s[j][v] - mnew);
        ps += s[j][v];
      }
#pragma unroll
      for (int off = 1; off < 16; off <<= 1) ps += __shfl_xor(ps, off);
      rowsum[v] = rowsum[v] * scl + ps;
#pragma unroll
      for (int dt = 0; dt < 4; ++dt) Oacc[dt][v] *= scl;
    }

    // transpose P from C-layout to A-layout through per-wave LDS strip
    _Float16* pb = &pbuf[wave][0];
#pragma unroll
    for (int v = 0; v < 8; ++v) {
      int row = grp * 8 + v;
#pragma unroll
      for (int j = 0; j < 4; ++j)
        pb[row * 72 + j * 16 + (lane & 15)] = (_Float16)s[j][v];
    }
    asm volatile("s_wait_dscnt 0" ::: "memory");
    v16h pa0 = load_A_lds(pb, 72, lane);        // P cols 0..31
    v16h pa1 = load_A_lds(pb + 32, 72, lane);   // P cols 32..63

    // O += P @ V : V chunk is 64 keys x 64 dims, channel-major => contiguous
#pragma unroll
    for (int dt = 0; dt < 4; ++dt) {
      v16h vb0 = load_B_fragT(Vp, MTOT, dt * 16, key0, lane);
      v16h vb1 = load_B_fragT(Vp, MTOT, dt * 16, key0 + 32, lane);
      Oacc[dt] = wmma_f16(pa0, vb0, Oacc[dt]);
      Oacc[dt] = wmma_f16(pa1, vb1, Oacc[dt]);
    }
  }

  // finalize: divide by rowsum, apply query mask, sum this wave's 16 rows
  float inv[8], mrow[8];
#pragma unroll
  for (int v = 0; v < 8; ++v) {
    float rs = rowsum[v];
    inv[v] = (rs > 1e-30f) ? 1.0f / rs : 0.0f;
    mrow[v] = maskrow[(size_t)b * SEQ_T + qw + grp * 8 + v];
  }
#pragma unroll
  for (int dt = 0; dt < 4; ++dt) {
    float c = 0.0f;
#pragma unroll
    for (int v = 0; v < 8; ++v) c += Oacc[dt][v] * inv[v] * mrow[v];
    atomicAdd(&sAcc[b * D_MODEL + h * HEAD_DIM + dt * 16 + (lane & 15)], c);
  }
}

// ---------------------------------------------------------------------------
// out[b,d] = (sAcc[b,:]/clip(den[b],1)) . Wo[d,:]   (tiny fp32 GEMM)
__global__ void final_proj(const float* __restrict__ sAcc,
                           const float* __restrict__ den,
                           const float* __restrict__ Wo,
                           float* __restrict__ out) {
  int idx = blockIdx.x * blockDim.x + threadIdx.x;  // 0..4095
  if (idx >= NBATCH * D_MODEL) return;
  int b = idx >> 10, d = idx & 1023;
  const float* s = sAcc + b * D_MODEL;
  const float* w = Wo + (size_t)d * D_MODEL;
  float acc = 0.0f;
  for (int k = 0; k < D_MODEL; ++k) acc += s[k] * w[k];
  out[idx] = acc / fmaxf(den[b], 1.0f);
}

// ---------------------------------------------------------------------------
extern "C" void kernel_launch(void* const* d_in, const int* in_sizes, int n_in,
                              void* d_out, int out_size, void* d_ws, size_t ws_size,
                              hipStream_t stream) {
  const float* x        = (const float*)d_in[0];
  const float* labels   = (const float*)d_in[1];
  const float* Wq       = (const float*)d_in[2];
  const float* Wk       = (const float*)d_in[3];
  const float* Wv       = (const float*)d_in[4];
  const float* Wo       = (const float*)d_in[5];
  const float* tau_ln_b = (const float*)d_in[7];
  const float* tau_w1   = (const float*)d_in[8];
  const float* tau_b1   = (const float*)d_in[9];
  const float* tau_w2   = (const float*)d_in[10];
  const float* tau_b2   = (const float*)d_in[11];
  const float* bias_ln_b= (const float*)d_in[13];
  const float* bias_w1  = (const float*)d_in[14];
  const float* bias_b1  = (const float*)d_in[15];
  const float* bias_w2  = (const float*)d_in[16];
  const float* bias_b2  = (const float*)d_in[17];
  const float* pos_embed= (const float*)d_in[18];
  float* out = (float*)d_out;

  char* w = (char*)d_ws;
  const size_t MB = 1ull << 20;
  _Float16* xh = (_Float16*)(w + 0);
  _Float16* Qh = (_Float16*)(w + 16 * MB);
  _Float16* Kh = (_Float16*)(w + 32 * MB);
  _Float16* Vt = (_Float16*)(w + 48 * MB);   // channel-major [D_MODEL][MTOT]
  _Float16* Wh = (_Float16*)(w + 64 * MB);
  float* sAcc    = (float*)(w + 66 * MB);
  float* den     = sAcc + NBATCH * D_MODEL;          // contiguous after sAcc
  float* qscale  = den + NBATCH;
  float* bscalar = qscale + NBATCH;
  float* maskrow = bscalar + NBATCH + 4;             // keep 16B alignment
  float* biaskey = maskrow + NBATCH * SEQ_T;

  const int ND = D_MODEL * D_MODEL;

  // 1) zero accumulators (sAcc + den, contiguous)
  zero_f32<<<(NBATCH * D_MODEL + NBATCH + 255) / 256, 256, 0, stream>>>(
      sAcc, NBATCH * D_MODEL + NBATCH);

  // 2) per-batch scalar MLPs
  prep_scalars<<<1, 32, 0, stream>>>(labels, tau_ln_b, tau_w1, tau_b1, tau_w2,
                                     tau_b2, bias_ln_b, bias_w1, bias_b1,
                                     bias_w2, bias_b2, qscale, bscalar);

  // 3) x -> f16, mask, key bias, den
  prep_rows<<<MTOT, 256, 0, stream>>>(x, xh, maskrow, biaskey, den, bscalar,
                                      pos_embed);

  // 4) Q/K/V projections (weight converted into reused f16 buffer)
  dim3 ggrid(MTOT / TILE_M, D_MODEL / TILE_N, 1);
  f2h<<<1024, 256, 0, stream>>>(Wq, Wh, ND);
  gemm_proj<<<ggrid, 256, 0, stream>>>(xh, Wh, Qh, D_MODEL, D_MODEL, qscale, 1, 0);
  f2h<<<1024, 256, 0, stream>>>(Wk, Wh, ND);
  gemm_proj<<<ggrid, 256, 0, stream>>>(xh, Wh, Kh, D_MODEL, D_MODEL, qscale, 0, 0);
  f2h<<<1024, 256, 0, stream>>>(Wv, Wh, ND);
  gemm_proj<<<ggrid, 256, 0, stream>>>(xh, Wh, Vt, D_MODEL, D_MODEL, qscale, 0, 1);

  // 5) fused flash attention + masked query-mean accumulation
  dim3 agrid(SEQ_T / 64, NHEAD, NBATCH);
  attn_fused<<<agrid, 128, 0, stream>>>(Qh, Kh, Vt, biaskey, maskrow, sAcc);

  // 6) collapsed output projection (mean over T commutes with Wo)
  final_proj<<<(NBATCH * D_MODEL + 255) / 256, 256, 0, stream>>>(sAcc, den, Wo,
                                                                 out);
  (void)in_sizes; (void)n_in; (void)out_size; (void)ws_size;
}
